// RowParallelLinear_fp8_51745765982774
// MI455X (gfx1250) — 8-bit matrix operand order repaired and checked by reading, compiled, not run on hardware
//
#include <hip/hip_runtime.h>

#define T_TOKENS 8192
#define K_DIM    4096
#define O_DIM    4096
#define KSCL     (K_DIM / 128)
#define NKSTEPS  (K_DIM / 128)

typedef __attribute__((ext_vector_type(16))) int   v16i;
typedef __attribute__((ext_vector_type(8)))  float v8f;

__device__ __forceinline__ void async_copy_b128(unsigned lds_off, const void* gptr) {
  asm volatile("global_load_async_to_lds_b128 %0, %1, off"
               :: "v"(lds_off), "v"(gptr)
               : "memory");
}
__device__ __forceinline__ void wait_asynccnt0() {
  asm volatile("s_wait_asynccnt 0x0" ::: "memory");
}

__device__ __forceinline__ unsigned int f32_to_e4m3_byte(float f) {
  unsigned int u    = __float_as_uint(f);
  unsigned int sign = (u >> 24) & 0x80u;
  unsigned int a    = u & 0x7fffffffu;
  if (a > 0x43e00000u) a = 0x43e00000u;
  if (a < 0x00800000u) return sign;
  int e = (int)(a >> 23) - 127;
  unsigned int m = (a & 0x7fffffu) | 0x800000u;
  int shift;
  if (e < -6) {
    shift = 20 + (-6 - e);
    if (shift > 24) return sign;
  } else {
    shift = 20;
  }
  unsigned int keep = m >> shift;
  unsigned int rem  = m & ((1u << shift) - 1u);
  unsigned int half = 1u << (shift - 1);
  keep += (rem > half || (rem == half && (keep & 1u))) ? 1u : 0u;
  unsigned int out;
  if (e < -6) {
    out = keep;
  } else {
    unsigned int ebits = (unsigned int)(e + 7);
    if (keep == 16u) { keep = 8u; ebits += 1u; }
    out = (ebits << 3) | (keep & 7u);
  }
  if (out > 0x7eu) out = 0x7eu;
  return sign | out;
}

__device__ __forceinline__ unsigned int pack4_e4m3(float a, float b, float c, float d) {
  return  f32_to_e4m3_byte(a)
       | (f32_to_e4m3_byte(b) << 8)
       | (f32_to_e4m3_byte(c) << 16)
       | (f32_to_e4m3_byte(d) << 24);
}

__global__ __launch_bounds__(256)
void quant_x_kernel(const float* __restrict__ x,
                    unsigned char* __restrict__ xq,
                    float* __restrict__ xscale) {
  __shared__ float red[256];
  const int row = blockIdx.x;
  const int tid = threadIdx.x;
  const float4* xr4 = (const float4*)(x + (size_t)row * K_DIM);

  float4 v[4];
  float amax = 0.0f;
#pragma unroll
  for (int j = 0; j < 4; ++j) {
    v[j] = xr4[tid * 4 + j];
    amax = fmaxf(amax, fmaxf(fmaxf(fabsf(v[j].x), fabsf(v[j].y)),
                             fmaxf(fabsf(v[j].z), fabsf(v[j].w))));
  }
  red[tid] = amax;
  __syncthreads();
#pragma unroll
  for (int s = 128; s > 0; s >>= 1) {
    if (tid < s) red[tid] = fmaxf(red[tid], red[tid + s]);
    __syncthreads();
  }
  const float am    = fmaxf(red[0], 1e-12f);
  const float scale = am * (1.0f / 448.0f);
  const float inv   = 448.0f / am;
  if (tid == 0) xscale[row] = scale;

  uint4 o;
  o.x = pack4_e4m3(v[0].x * inv, v[0].y * inv, v[0].z * inv, v[0].w * inv);
  o.y = pack4_e4m3(v[1].x * inv, v[1].y * inv, v[1].z * inv, v[1].w * inv);
  o.z = pack4_e4m3(v[2].x * inv, v[2].y * inv, v[2].z * inv, v[2].w * inv);
  o.w = pack4_e4m3(v[3].x * inv, v[3].y * inv, v[3].z * inv, v[3].w * inv);
  ((uint4*)(xq + (size_t)row * K_DIM))[tid] = o;
}

__global__ __launch_bounds__(256)
void quant_w_kernel(const float* __restrict__ w, unsigned char* __restrict__ wq) {
  const size_t idx = (size_t)blockIdx.x * blockDim.x + threadIdx.x;
  const float4* w4 = (const float4*)w;
  uint4 o;
  float4 a = w4[idx * 4 + 0];
  float4 b = w4[idx * 4 + 1];
  float4 c = w4[idx * 4 + 2];
  float4 d = w4[idx * 4 + 3];
  o.x = pack4_e4m3(a.x, a.y, a.z, a.w);
  o.y = pack4_e4m3(b.x, b.y, b.z, b.w);
  o.z = pack4_e4m3(c.x, c.y, c.z, c.w);
  o.w = pack4_e4m3(d.x, d.y, d.z, d.w);
  ((uint4*)wq)[idx] = o;
}

#define TILE_BYTES (128 * 128)

__global__ __launch_bounds__(256)
void gemm_fp8_kernel(const unsigned char* __restrict__ xq,
                     const unsigned char* __restrict__ wq,
                     const float* __restrict__ xscale,
                     const float* __restrict__ wsinv,
                     float* __restrict__ y) {
  __shared__ __align__(16) unsigned char sA[2][TILE_BYTES];
  __shared__ __align__(16) unsigned char sB[2][TILE_BYTES];

  const int tid  = threadIdx.x;
  const int lane = tid & 31;
  const int wave = tid >> 5;
  const int wy   = wave >> 1;
  const int wx   = wave & 1;
  const int ml   = lane & 15;
  const int hi   = lane >> 4;

  const int m_block = blockIdx.y * 128;
  const int o_block = blockIdx.x;
  const int o_base  = o_block * 128;

  unsigned ldsA[4], ldsB[4];
  const unsigned char* gA[4];
  const unsigned char* gB[4];
#pragma unroll
  for (int r = 0; r < 4; ++r) {
    const int idx = r * 256 + tid;
    const int row = idx >> 3;
    const int col = (idx & 7) * 16;
    const int csw = col ^ ((row & 7) << 4);
    ldsA[r] = (unsigned)(unsigned long long)&sA[0][row * 128 + csw];
    ldsB[r] = (unsigned)(unsigned long long)&sB[0][row * 128 + csw];
    gA[r]   = xq + (size_t)(m_block + row) * K_DIM + col;
    gB[r]   = wq + (size_t)(o_base + row) * K_DIM + col;
  }

  v8f acc[2][4];
#pragma unroll
  for (int a2 = 0; a2 < 2; ++a2)
#pragma unroll
    for (int t = 0; t < 4; ++t)
      acc[a2][t] = (v8f){0.f, 0.f, 0.f, 0.f, 0.f, 0.f, 0.f, 0.f};

#pragma unroll
  for (int r = 0; r < 4; ++r) async_copy_b128(ldsA[r], gA[r]);
#pragma unroll
  for (int r = 0; r < 4; ++r) async_copy_b128(ldsB[r], gB[r]);

  for (int i = 0; i < NKSTEPS; ++i) {
    const int cur = i & 1;
    wait_asynccnt0();
    __syncthreads();

    if (i + 1 < NKSTEPS) {
      const unsigned bo = (unsigned)((1 - cur) * TILE_BYTES);
      const int kk = (i + 1) * 128;
#pragma unroll
      for (int r = 0; r < 4; ++r) async_copy_b128(ldsA[r] + bo, gA[r] + kk);
#pragma unroll
      for (int r = 0; r < 4; ++r) async_copy_b128(ldsB[r] + bo, gB[r] + kk);
    }

    const float s = wsinv[o_block * KSCL + i];

    v16i Afr[2];
#pragma unroll
    for (int a2 = 0; a2 < 2; ++a2) {
      const int ra = wy * 32 + a2 * 16 + ml;
      const unsigned char* abase = &sA[cur][ra * 128 + 8 * hi];
      const unsigned sw = (unsigned)((ra & 7) << 4);
#pragma unroll
      for (int j = 0; j < 8; ++j) {
        int2 d = *(const int2*)(abase + ((unsigned)(16 * j) ^ sw));
        Afr[a2][2 * j]     = d.x;
        Afr[a2][2 * j + 1] = d.y;
      }
    }

#pragma unroll
    for (int t = 0; t < 4; ++t) {
      const int rb = wx * 64 + t * 16 + ml;
      const unsigned char* bbase = &sB[cur][rb * 128 + 8 * hi];
      const unsigned swb = (unsigned)((rb & 7) << 4);
      v16i B;
#pragma unroll
      for (int q = 0; q < 8; ++q) {
        int2 d = *(const int2*)(bbase + ((unsigned)(16 * q) ^ swb));
        B[2 * q]     = d.x;
        B[2 * q + 1] = d.y;
      }
#pragma unroll
      for (int a2 = 0; a2 < 2; ++a2) {
        v8f zero = (v8f){0.f, 0.f, 0.f, 0.f, 0.f, 0.f, 0.f, 0.f};
        v8f pr = __builtin_amdgcn_wmma_f32_16x16x128_fp8_fp8(
            Afr[a2], B, (short)0, zero,  false,  false);
#pragma unroll
        for (int rr = 0; rr < 8; ++rr)
          acc[a2][t][rr] = fmaf(s, pr[rr], acc[a2][t][rr]);
      }
    }
  }

#pragma unroll
  for (int a2 = 0; a2 < 2; ++a2) {
#pragma unroll
    for (int rr = 0; rr < 8; ++rr) {
      const int row  = m_block + wy * 32 + a2 * 16 + rr + 8 * hi;
      const float xs = xscale[row];
      float* yr = y + (size_t)row * O_DIM + (o_base + wx * 64 + ml);
      yr[0]  = acc[a2][0][rr] * xs;
      yr[16] = acc[a2][1][rr] * xs;
      yr[32] = acc[a2][2][rr] * xs;
      yr[48] = acc[a2][3][rr] * xs;
    }
  }
}

extern "C" void kernel_launch(void* const* d_in, const int* in_sizes, int n_in,
                              void* d_out, int out_size, void* d_ws, size_t ws_size,
                              hipStream_t stream) {
  (void)in_sizes; (void)n_in; (void)out_size; (void)ws_size;
  const float* x     = (const float*)d_in[0];
  const float* w     = (const float*)d_in[1];
  const float* wsinv = (const float*)d_in[2];
  float* y           = (float*)d_out;

  unsigned char* ws = (unsigned char*)d_ws;
  unsigned char* xq = ws;
  unsigned char* wq = ws + (size_t)T_TOKENS * K_DIM;
  float* xscale     = (float*)(wq + (size_t)O_DIM * K_DIM);

  quant_x_kernel<<<T_TOKENS, 256, 0, stream>>>(x, xq, xscale);
  quant_w_kernel<<<(O_DIM * (size_t)K_DIM) / (256 * 16), 256, 0, stream>>>(w, wq);

  dim3 grid(O_DIM / 128, T_TOKENS / 128);
  gemm_fp8_kernel<<<grid, 256, 0, stream>>>(xq, wq, xscale, wsinv, y);
}
